// VisionTransformerBlock_17927193493613
// MI455X (gfx1250) — matrix operand order repaired and checked by reading, not compiled, not run on hardware
//
#include <hip/hip_runtime.h>

typedef __attribute__((ext_vector_type(16))) __bf16 v16bf;
typedef __attribute__((ext_vector_type(8)))  __bf16 v8bf;
typedef __attribute__((ext_vector_type(8)))  float  v8f;

#define DEVI __device__ __forceinline__

constexpr int NH   = 16;    // heads
constexpr int NTOK = 1024;  // tokens (32x32)
constexpr int DIMC = 768;   // channels
constexpr int HD   = 48;    // head dim
constexpr int HDP  = 64;    // padded head dim (K multiple of 32)
constexpr float SCALE_F = 0.14433756729740643f; // 48^-0.5

// ---------- helpers ----------
DEVI unsigned short f2bf(float f) {
  union { float f; unsigned u; } c; c.f = f;
  unsigned u = c.u;
  unsigned r = (u + 0x7FFFu + ((u >> 16) & 1u)) >> 16; // RNE
  return (unsigned short)r;
}

DEVI v16bf cat8(v8bf lo, v8bf hi) {
  return __builtin_shufflevector(lo, hi, 0,1,2,3,4,5,6,7,8,9,10,11,12,13,14,15);
}

// A-matrix 16x32 bf16 fragment (wave32 ISA layout):
// lane L: row = r0 + (L&15); half = L>>4
// VGPR0-3: K = k0 + half*8 + 0..7 ; VGPR4-7: K = k0 + 16 + half*8 + 0..7
DEVI v16bf load_a(const unsigned short* p, int ld, int r0, int k0, int lane) {
  const unsigned short* q = p + (size_t)(r0 + (lane & 15)) * ld + k0 + (lane >> 4) * 8;
  v8bf lo = *(const v8bf*)q;
  v8bf hi = *(const v8bf*)(q + 16);
  return cat8(lo, hi);
}

// B-matrix 32x16 bf16 fragment, B stored as [Nrows][K] row-major (row = output col):
// lane L: N = n0 + (L&15); K = k0 + (L>>4)*8 + 0..7, then k0 + 16 + (L>>4)*8 + 0..7
DEVI v16bf load_b(const unsigned short* p, int ld, int n0, int k0, int lane) {
  const unsigned short* q = p + (size_t)(n0 + (lane & 15)) * ld + k0 + (lane >> 4) * 8;
  v8bf lo = *(const v8bf*)q;
  v8bf hi = *(const v8bf*)(q + 16);
  return cat8(lo, hi);
}

DEVI v8f wmma_bf16(v16bf a, v16bf b, v8f c) {
  return __builtin_amdgcn_wmma_f32_16x16x32_bf16(false, a, false, b, (short)0, c, false, false);
}

// ---------- prep kernels ----------
__global__ void convert_kernel(const float* __restrict__ src,
                               unsigned short* __restrict__ dst, int n) {
  int i = blockIdx.x * blockDim.x + threadIdx.x;
  if (i < n) dst[i] = f2bf(src[i]);
}

__global__ void zero_kernel(unsigned int* __restrict__ dst, int nwords) {
  int i = blockIdx.x * blockDim.x + threadIdx.x;
  if (i < nwords) dst[i] = 0u;
}

// ---------- positional softmax: pos_sm[h, n, m] (batch independent) ----------
__global__ void pos_kernel(const float* __restrict__ Wpos,
                           const float* __restrict__ bpos,
                           unsigned short* __restrict__ posP) {
  int wv = threadIdx.x >> 5, lane = threadIdx.x & 31;
  int row = blockIdx.x * 8 + wv;          // 16*1024 rows
  int h = row >> 10, n = row & 1023;
  float w0 = Wpos[h*3+0], w1 = Wpos[h*3+1], w2 = Wpos[h*3+2], bb = bpos[h];
  int nx = n & 31, ny = n >> 5;
  float sv[32];
  float mx = -1e30f;
#pragma unroll
  for (int j = 0; j < 32; ++j) {
    int m = lane + 32 * j;
    float dx = (float)((m & 31) - nx);
    float dy = (float)((m >> 5) - ny);
    float s = w0 * dx + w1 * dy + w2 * (dx * dx + dy * dy) + bb;
    sv[j] = s;
    mx = fmaxf(mx, s);
  }
#pragma unroll
  for (int off = 1; off < 32; off <<= 1) mx = fmaxf(mx, __shfl_xor(mx, off, 32));
  float sm = 0.f;
#pragma unroll
  for (int j = 0; j < 32; ++j) { sv[j] = __expf(sv[j] - mx); sm += sv[j]; }
#pragma unroll
  for (int off = 1; off < 32; off <<= 1) sm += __shfl_xor(sm, off, 32);
  float inv = 1.f / sm;
  unsigned short* dst = posP + (size_t)row * NTOK;
#pragma unroll
  for (int j = 0; j < 32; ++j) dst[lane + 32 * j] = f2bf(sv[j] * inv);
}

// ---------- fused QKV GEMM, 2x4 register-blocked ----------
// Each wave: 32 rows x 64 cols of the [4096 x 2304] concat output.
// Per K-chunk: 2 A frags + 4 B frags -> 8 WMMAs (0.75KB L2 traffic per WMMA).
__global__ void qkv_kernel(const unsigned short* __restrict__ xbf,
                           const unsigned short* __restrict__ Wqbf,
                           const unsigned short* __restrict__ Wkbf,
                           const unsigned short* __restrict__ Wvbf,
                           unsigned short* __restrict__ qbf,
                           unsigned short* __restrict__ kbf,
                           unsigned short* __restrict__ vT) {
  int wv = threadIdx.x >> 5, lane = threadIdx.x & 31;
  int job = blockIdx.x * 8 + wv;          // 128 row-blocks * 36 col-blocks
  int mb = job & 127;                     // 32-row block
  int nb = job >> 7;                      // 0..35 : 64-col block of concat out
  int mat = nb / 12;                      // 12 col-blocks per weight matrix
  int cc0 = (nb % 12) * 64;
  const unsigned short* Wb = (mat == 0) ? Wqbf : (mat == 1) ? Wkbf : Wvbf;
  int m0 = mb * 32;

  v8f acc[2][4] = {};
  for (int k0 = 0; k0 < DIMC; k0 += 32) {
    v16bf a0 = load_a(xbf, DIMC, m0,      k0, lane);
    v16bf a1 = load_a(xbf, DIMC, m0 + 16, k0, lane);
#pragma unroll
    for (int j = 0; j < 4; ++j) {
      v16bf b = load_b(Wb, DIMC, cc0 + j * 16, k0, lane);
      acc[0][j] = wmma_bf16(a0, b, acc[0][j]);
      acc[1][j] = wmma_bf16(a1, b, acc[1][j]);
    }
  }

  int half = lane >> 4;
#pragma unroll
  for (int j = 0; j < 4; ++j) {
    int c = cc0 + j * 16 + (lane & 15);
    int h = c / HD, d = c % HD;
#pragma unroll
    for (int i = 0; i < 2; ++i) {
#pragma unroll
      for (int r = 0; r < 8; ++r) {
        int m = m0 + i * 16 + r + 8 * half;    // global row: b*1024 + n
        int bidx = m >> 10, n = m & 1023;
        int bh = bidx * NH + h;
        float val = acc[i][j][r];
        if (mat == 0)      qbf[((size_t)bh * NTOK + n) * HDP + d] = f2bf(val * SCALE_F);
        else if (mat == 1) kbf[((size_t)bh * NTOK + n) * HDP + d] = f2bf(val);
        else               vT[((size_t)bh * HDP + d) * NTOK + n]  = f2bf(val); // transposed
      }
    }
  }
}

// ---------- attention: out_h = (1-g)*softmax(qk)·V + g*pos_sm·V ----------
__global__ void attn_kernel(const unsigned short* __restrict__ qbf,
                            const unsigned short* __restrict__ kbf,
                            const unsigned short* __restrict__ vT,
                            const unsigned short* __restrict__ posP,
                            const float* __restrict__ gate,
                            unsigned short* __restrict__ ohead) {
  __shared__ __align__(16) unsigned short Plds[8][16][40]; // per-wave D->A transpose staging
  int wv = threadIdx.x >> 5, lane = threadIdx.x & 31;
  int w = blockIdx.x * 8 + wv;            // 4 * 16 * 64 waves
  int b = w >> 10;
  int h = (w >> 6) & 15;
  int n0 = (w & 63) * 16;
  int half = lane >> 4;
  int bh = b * NH + h;
  const unsigned short* qb = qbf + (size_t)bh * NTOK * HDP;
  const unsigned short* kb = kbf + (size_t)bh * NTOK * HDP;
  const unsigned short* vb = vT  + (size_t)bh * HDP * NTOK;
  const unsigned short* pp = posP + (size_t)h * NTOK * NTOK;

  v16bf qa0 = load_a(qb, HDP, n0, 0, lane);
  v16bf qa1 = load_a(qb, HDP, n0, 32, lane);

  // ---- pass 1: row max / row sum (online) ----
  float rm[8], rs[8];
#pragma unroll
  for (int r = 0; r < 8; ++r) { rm[r] = -1e30f; rs[r] = 0.f; }

  for (int m0 = 0; m0 < NTOK; m0 += 32) {
    v8f z = {};
    v8f s0 = wmma_bf16(qa0, load_b(kb, HDP, m0, 0, lane), z);
    s0 = wmma_bf16(qa1, load_b(kb, HDP, m0, 32, lane), s0);
    v8f s1 = wmma_bf16(qa0, load_b(kb, HDP, m0 + 16, 0, lane), z);
    s1 = wmma_bf16(qa1, load_b(kb, HDP, m0 + 16, 32, lane), s1);
#pragma unroll
    for (int r = 0; r < 8; ++r) {
      float cm = fmaxf(s0[r], s1[r]);
#pragma unroll
      for (int off = 1; off < 16; off <<= 1) cm = fmaxf(cm, __shfl_xor(cm, off, 32));
      float nm = fmaxf(rm[r], cm);
      float cs = __expf(s0[r] - nm) + __expf(s1[r] - nm);
#pragma unroll
      for (int off = 1; off < 16; off <<= 1) cs += __shfl_xor(cs, off, 32);
      rs[r] = rs[r] * __expf(rm[r] - nm) + cs;
      rm[r] = nm;
    }
  }
  float rinv[8];
#pragma unroll
  for (int r = 0; r < 8; ++r) rinv[r] = 1.f / rs[r];

  // ---- pass 2: O1 = P_patch·V, O2 = P_pos·V (shared V fragments) ----
  v8f o1[3] = {}, o2[3] = {};
  for (int m0 = 0; m0 < NTOK; m0 += 32) {
    v8f z = {};
    v8f s0 = wmma_bf16(qa0, load_b(kb, HDP, m0, 0, lane), z);
    s0 = wmma_bf16(qa1, load_b(kb, HDP, m0, 32, lane), s0);
    v8f s1 = wmma_bf16(qa0, load_b(kb, HDP, m0 + 16, 0, lane), z);
    s1 = wmma_bf16(qa1, load_b(kb, HDP, m0 + 16, 32, lane), s1);
#pragma unroll
    for (int r = 0; r < 8; ++r) {
      int rr = r + 8 * half;              // accumulator D layout -> LDS row/col
      Plds[wv][rr][lane & 15]        = f2bf(__expf(s0[r] - rm[r]) * rinv[r]);
      Plds[wv][rr][16 + (lane & 15)] = f2bf(__expf(s1[r] - rm[r]) * rinv[r]);
    }
    asm volatile("s_wait_dscnt 0" ::: "memory");
    const unsigned short* la = &Plds[wv][0][0] + (lane & 15) * 40 + half * 8;
    v16bf pa = cat8(*(const v8bf*)la, *(const v8bf*)(la + 16));   // A-layout P
    v16bf ga = load_a(pp, NTOK, n0, m0, lane);                    // pos_sm A frag
#pragma unroll
    for (int t = 0; t < 3; ++t) {
      v16bf vf = load_b(vb, NTOK, t * 16, m0, lane);              // V^T rows = d
      o1[t] = wmma_bf16(pa, vf, o1[t]);
      o2[t] = wmma_bf16(ga, vf, o2[t]);
    }
  }

  float g = 1.f / (1.f + __expf(-gate[h]));   // sigmoid gate; renorm is identity
#pragma unroll
  for (int t = 0; t < 3; ++t) {
#pragma unroll
    for (int r = 0; r < 8; ++r) {
      int row = b * NTOK + n0 + r + 8 * half;
      int col = h * HD + t * 16 + (lane & 15);
      ohead[(size_t)row * DIMC + col] = f2bf((1.f - g) * o1[t][r] + g * o2[t][r]);
    }
  }
}

// ---------- final projection, 2x4 register-blocked: out = ohead @ Wproj.T + bproj ----------
__global__ void proj_kernel(const unsigned short* __restrict__ ohead,
                            const unsigned short* __restrict__ Wpjbf,
                            const float* __restrict__ bproj,
                            float* __restrict__ out) {
  int wv = threadIdx.x >> 5, lane = threadIdx.x & 31;
  int job = blockIdx.x * 8 + wv;          // 128 row-blocks * 12 col-blocks
  int mb = job & 127;
  int nb = job >> 7;                      // 0..11
  int m0 = mb * 32, cc0 = nb * 64;

  v8f acc[2][4] = {};
  for (int k0 = 0; k0 < DIMC; k0 += 32) {
    v16bf a0 = load_a(ohead, DIMC, m0,      k0, lane);
    v16bf a1 = load_a(ohead, DIMC, m0 + 16, k0, lane);
#pragma unroll
    for (int j = 0; j < 4; ++j) {
      v16bf bfr = load_b(Wpjbf, DIMC, cc0 + j * 16, k0, lane);
      acc[0][j] = wmma_bf16(a0, bfr, acc[0][j]);
      acc[1][j] = wmma_bf16(a1, bfr, acc[1][j]);
    }
  }

  int half = lane >> 4;
#pragma unroll
  for (int j = 0; j < 4; ++j) {
    int c = cc0 + j * 16 + (lane & 15);
    float bias = bproj[c];
#pragma unroll
    for (int i = 0; i < 2; ++i) {
#pragma unroll
      for (int r = 0; r < 8; ++r)
        out[(size_t)(m0 + i * 16 + r + 8 * half) * DIMC + c] = acc[i][j][r] + bias;
    }
  }
}

// ---------- launch ----------
extern "C" void kernel_launch(void* const* d_in, const int* in_sizes, int n_in,
                              void* d_out, int out_size, void* d_ws, size_t ws_size,
                              hipStream_t stream) {
  const float* x     = (const float*)d_in[0];
  const float* Wq    = (const float*)d_in[1];
  const float* Wk    = (const float*)d_in[2];
  const float* Wv    = (const float*)d_in[3];
  const float* Wpos  = (const float*)d_in[4];
  const float* bpos  = (const float*)d_in[5];
  const float* gate  = (const float*)d_in[6];
  const float* Wproj = (const float*)d_in[7];
  const float* bproj = (const float*)d_in[8];
  float* out = (float*)d_out;

  const size_t XN  = (size_t)4 * NTOK * DIMC;        // 3,145,728
  const size_t WN  = (size_t)DIMC * DIMC;            // 589,824
  const size_t QKN = (size_t)4 * NH * NTOK * HDP;    // 4,194,304
  const size_t PPN = (size_t)NH * NTOK * NTOK;       // 16,777,216

  char* p = (char*)d_ws;
  unsigned short* xbf   = (unsigned short*)p; p += XN * 2;
  unsigned short* Wqbf  = (unsigned short*)p; p += WN * 2;
  unsigned short* Wkbf  = (unsigned short*)p; p += WN * 2;
  unsigned short* Wvbf  = (unsigned short*)p; p += WN * 2;
  unsigned short* Wpjbf = (unsigned short*)p; p += WN * 2;
  unsigned short* qbf   = (unsigned short*)p; p += QKN * 2;   // q, k, vT contiguous
  unsigned short* kbf   = (unsigned short*)p; p += QKN * 2;
  unsigned short* vT    = (unsigned short*)p; p += QKN * 2;
  unsigned short* posP  = (unsigned short*)p; p += PPN * 2;
  unsigned short* ohead = (unsigned short*)p; p += XN * 2;

  convert_kernel<<<(int)((XN + 255) / 256), 256, 0, stream>>>(x, xbf, (int)XN);
  convert_kernel<<<(int)((WN + 255) / 256), 256, 0, stream>>>(Wq, Wqbf, (int)WN);
  convert_kernel<<<(int)((WN + 255) / 256), 256, 0, stream>>>(Wk, Wkbf, (int)WN);
  convert_kernel<<<(int)((WN + 255) / 256), 256, 0, stream>>>(Wv, Wvbf, (int)WN);
  convert_kernel<<<(int)((WN + 255) / 256), 256, 0, stream>>>(Wproj, Wpjbf, (int)WN);

  int zw = (int)(3 * QKN / 2); // q,k,vT zero-padded in one shot (u32 words)
  zero_kernel<<<(zw + 255) / 256, 256, 0, stream>>>((unsigned*)qbf, zw);

  pos_kernel<<<2048, 256, 0, stream>>>(Wpos, bpos, posP);            // 16384 waves
  qkv_kernel<<<576, 256, 0, stream>>>(xbf, Wqbf, Wkbf, Wvbf, qbf, kbf, vT);  // 4608 wave-jobs
  attn_kernel<<<512, 256, 0, stream>>>(qbf, kbf, vT, posP, gate, ohead);     // 4096 wave-jobs
  proj_kernel<<<192, 256, 0, stream>>>(ohead, Wpjbf, bproj, out);            // 1536 wave-jobs
}
